// LocalBandNoSimBlock_81801947120113
// MI455X (gfx1250) — compile-verified, matrix operand order checked by reading, not run on hardware
//
#include <hip/hip_runtime.h>
#include <hip/hip_bf16.h>

typedef unsigned short u16;
typedef __attribute__((ext_vector_type(16))) __bf16        v16bf;
typedef __attribute__((ext_vector_type(8)))  float         v8f;
typedef __attribute__((ext_vector_type(4)))  unsigned int  u32x4;
typedef __attribute__((ext_vector_type(8)))  unsigned int  u32x8;

union Frag { uint4 u[2]; v16bf v; };

#define TOKENS 9216
#define DMODEL 512
#define GSIDE  96
#define BK     64          // K-chunk per LDS stage (2 WMMA k-steps)

__device__ __forceinline__ u16 f32_to_bf16(float f) {
    unsigned int u = __float_as_uint(f);
    u += 0x7FFFu + ((u >> 16) & 1u);           // round-to-nearest-even
    return (u16)(u >> 16);
}

__device__ __forceinline__ float gelu_exact(float x) {
    return 0.5f * x * (1.0f + erff(x * 0.70710678118654752f));
}

// ---------------------------------------------------------------------------
// Tensor Data Mover: async DMA of a 2-D bf16 tile (tile_rows x tile_k) from a
// row-major [rows_total, K_len] tensor in global memory into LDS (row-major,
// row stride = tile_k elements).  D# built per CDNA5 ISA ch.8:
//   group0: count=1 | lds_addr | global_addr[56:0] | type=2
//   group1: data_size=1(2B) | tensor_dim0=K | tensor_dim1=rows |
//           tile_dim0=tile_k | tile_dim1=tile_rows | dim0_stride=K
// Tracked with TENSORcnt (s_wait_tensorcnt).
// ---------------------------------------------------------------------------
__device__ __forceinline__ void tdm_load_tile_2d(unsigned lds_addr, const u16* gsrc,
                                                 unsigned K_len, unsigned rows_total,
                                                 unsigned tile_k, unsigned tile_rows) {
    const unsigned long long ga = (unsigned long long)(uintptr_t)gsrc;
    u32x4 g0;
    g0[0] = 1u;                                        // count=1, user D#
    g0[1] = lds_addr;                                  // LDS byte address
    g0[2] = (unsigned)ga;                              // global_addr[31:0]
    g0[3] = ((unsigned)(ga >> 32) & 0x01FFFFFFu)       // global_addr[56:32]
            | 0x80000000u;                             // type=2 ("image")
    u32x8 g1;
    g1[0] = (1u << 16);                                // data_size=1 -> 2 bytes
    g1[1] = (K_len & 0xFFFFu) << 16;                   // tensor_dim0[15:0]
    g1[2] = (K_len >> 16) | ((rows_total & 0xFFFFu) << 16);  // dim0 hi | dim1 lo
    g1[3] = (rows_total >> 16) | (tile_k << 16);       // dim1 hi | tile_dim0
    g1[4] = tile_rows;                                 // tile_dim1 (tile_dim2=0)
    g1[5] = K_len;                                     // tensor_dim0_stride[31:0]
    g1[6] = 0u;                                        // stride hi | dim1_stride lo
    g1[7] = 0u;
    asm volatile("tensor_load_to_lds %0, %1" :: "s"(g0), "s"(g1) : "memory");
}

// ---------------------------------------------------------------------------
// LayerNorm (D=512) -> bf16 output.  One block (256 thr) per row.
// ---------------------------------------------------------------------------
__global__ __launch_bounds__(256)
void ln_bf16_kernel(const float* __restrict__ x, const float* __restrict__ g,
                    const float* __restrict__ b, u16* __restrict__ out) {
    const int r   = blockIdx.x;
    const int tid = threadIdx.x;
    const float v0 = x[(size_t)r * DMODEL + tid];
    const float v1 = x[(size_t)r * DMODEL + tid + 256];
    float s  = v0 + v1;
    float s2 = v0 * v0 + v1 * v1;
    #pragma unroll
    for (int o = 16; o > 0; o >>= 1) {
        s  += __shfl_xor(s,  o, 32);
        s2 += __shfl_xor(s2, o, 32);
    }
    __shared__ float rs[8], rs2[8];
    if ((tid & 31) == 0) { rs[tid >> 5] = s; rs2[tid >> 5] = s2; }
    __syncthreads();
    float ts = 0.f, ts2 = 0.f;
    #pragma unroll
    for (int i = 0; i < 8; ++i) { ts += rs[i]; ts2 += rs2[i]; }
    const float mean = ts * (1.0f / DMODEL);
    const float var  = ts2 * (1.0f / DMODEL) - mean * mean;
    const float rstd = rsqrtf(var + 1e-5f);
    out[(size_t)r * DMODEL + tid]       = f32_to_bf16((v0 - mean) * rstd * g[tid]       + b[tid]);
    out[(size_t)r * DMODEL + tid + 256] = f32_to_bf16((v1 - mean) * rstd * g[tid + 256] + b[tid + 256]);
}

// ---------------------------------------------------------------------------
// Weight convert f32[K,N] -> bf16 transposed [N,K] (so B-tiles are row-per-N).
// ---------------------------------------------------------------------------
__global__ __launch_bounds__(256)
void cvt_t_kernel(const float* __restrict__ W, u16* __restrict__ Wt, int K, int N) {
    const int e = blockIdx.x * 256 + threadIdx.x;
    if (e >= K * N) return;
    const int k = e / N, n = e % N;
    Wt[(size_t)n * K + k] = f32_to_bf16(W[e]);
}

// ---------------------------------------------------------------------------
// bf16 WMMA GEMM with TDM double-buffered tile staging.
// C[M,Nout] = epilogue( A[M,K] @ Bt[Nout,K]^T + bias )
// Block = 256 thr (8 wave32), tile = 128x64, K-chunk 64 (2 k-steps / stage).
// Wave 0 drives the Tensor Data Mover; all waves compute 8 WMMAs per stage.
// ---------------------------------------------------------------------------
__global__ __launch_bounds__(256)
void gemm_bf16_kernel(const u16* __restrict__ A, const u16* __restrict__ Bt,
                      const float* __restrict__ bias, const float* __restrict__ res,
                      float* __restrict__ out_f32, u16* __restrict__ out_bf16,
                      int M, int Nout, int K, int gelu_flag) {
    __shared__ __align__(16) u16 As[2][128 * BK];   // 2 x 16 KB
    __shared__ __align__(16) u16 Bs[2][64 * BK];    // 2 x  8 KB

    const int tid   = threadIdx.x;
    const int w     = tid >> 5;
    const int lane  = tid & 31;
    const int lm    = lane & 15;     // M (A) / N (B) index within fragment
    const int khalf = lane >> 4;     // which 8-wide K-half this lane holds
    const int bm    = blockIdx.x;    // 128-row tile
    const int bn    = blockIdx.y;    // 64-col tile

    v8f c[4];
    #pragma unroll
    for (int i = 0; i < 4; ++i) c[i] = (v8f){0.f,0.f,0.f,0.f,0.f,0.f,0.f,0.f};

    const int   steps = K / BK;
    const u16*  Abase = A  + (size_t)bm * 128 * K;
    const u16*  Bbase = Bt + (size_t)bn * 64  * K;
    const bool  issuer = (tid < 32);                 // wave 0 drives the TDM

    if (issuer) {
        tdm_load_tile_2d((unsigned)(uintptr_t)&As[0][0], Abase, K, (unsigned)M,   BK, 128);
        tdm_load_tile_2d((unsigned)(uintptr_t)&Bs[0][0], Bbase, K, (unsigned)Nout, BK, 64);
    }

    for (int i = 0; i < steps; ++i) {
        const int cur = i & 1;
        if (issuer) {
            if (i + 1 < steps) {
                const int nxt = cur ^ 1;
                // safe: end-of-iteration barrier of step i-1 guarantees no wave
                // still reads buffer `nxt`
                tdm_load_tile_2d((unsigned)(uintptr_t)&As[nxt][0],
                                 Abase + (size_t)(i + 1) * BK, K, (unsigned)M, BK, 128);
                tdm_load_tile_2d((unsigned)(uintptr_t)&Bs[nxt][0],
                                 Bbase + (size_t)(i + 1) * BK, K, (unsigned)Nout, BK, 64);
                // TENSORcnt is in-order: <=2 outstanding means current tile done
                __builtin_amdgcn_s_wait_tensorcnt(2);
            } else {
                __builtin_amdgcn_s_wait_tensorcnt(0);
            }
        }
        __syncthreads();                             // current tile visible to all

        const u16* Ab = &As[cur][0];
        const u16* Bb = &Bs[cur][0];
        #pragma unroll
        for (int j = 0; j < 2; ++j) {                // two 32-wide k-steps
            Frag a;
            const int abase = (w * 16 + lm) * BK + j * 32 + khalf * 8;
            a.u[0] = *(const uint4*)&Ab[abase];
            a.u[1] = *(const uint4*)&Ab[abase + 16];
            #pragma unroll
            for (int nb = 0; nb < 4; ++nb) {
                Frag bfrag;
                const int bbase = (nb * 16 + lm) * BK + j * 32 + khalf * 8;
                bfrag.u[0] = *(const uint4*)&Bb[bbase];
                bfrag.u[1] = *(const uint4*)&Bb[bbase + 16];
                c[nb] = __builtin_amdgcn_wmma_f32_16x16x32_bf16(
                            false, a.v, false, bfrag.v, (short)0, c[nb], false, false);
            }
        }
        __syncthreads();                             // all reads of `cur` done
    }

    // ---- epilogue: lane 0-15 -> N=lane, rows M=i; lane 16-31 -> M=8+i ----
    const int rowbase = bm * 128 + w * 16 + (khalf << 3);
    const int colbase = bn * 64;
    #pragma unroll
    for (int nb = 0; nb < 4; ++nb) {
        const int col = colbase + nb * 16 + lm;
        const float bv = bias ? bias[col] : 0.0f;
        #pragma unroll
        for (int i = 0; i < 8; ++i) {
            const int row = rowbase + i;
            float v = c[nb][i] + bv;
            if (gelu_flag) v = gelu_exact(v);
            if (res)      v += res[(size_t)row * Nout + col];
            if (out_f32)  out_f32[(size_t)row * Nout + col] = v;
            if (out_bf16) out_bf16[(size_t)row * Nout + col] = f32_to_bf16(v);
        }
    }
}

// ---------------------------------------------------------------------------
// 24-neighbor local attention.  One block (256 thr = 8 waves) per token.
// Each wave dots one neighbor (3 rounds); softmax; weighted V gather.
// ---------------------------------------------------------------------------
__global__ __launch_bounds__(256)
void attn24_kernel(const float* __restrict__ q, const float* __restrict__ k,
                   const float* __restrict__ v, const int* __restrict__ grid,
                   u16* __restrict__ out) {
    __shared__ float qs[DMODEL];
    __shared__ float sc[24];
    __shared__ float probs[24];
    __shared__ int   nidx[24];

    const int n   = blockIdx.x;
    const int tid = threadIdx.x;
    qs[tid]       = q[(size_t)n * DMODEL + tid];
    qs[tid + 256] = q[(size_t)n * DMODEL + tid + 256];
    const int gx = grid[2 * n + 0];
    const int gy = grid[2 * n + 1];
    __syncthreads();

    const int w = tid >> 5, lane = tid & 31;
    for (int kk = w; kk < 24; kk += 8) {
        const int lin = kk + (kk >= 12 ? 1 : 0);     // skip (0,0) at linear 12
        const int dx = lin / 5 - 2, dy = lin % 5 - 2;
        const int nx = gx + dx, ny = gy + dy;
        const bool valid = (nx >= 0) & (nx < GSIDE) & (ny >= 0) & (ny < GSIDE);
        const int idx = valid ? nx * GSIDE + ny : 0;
        const float* kr = k + (size_t)idx * DMODEL;
        float p = 0.f;
        #pragma unroll
        for (int j = 0; j < DMODEL / 32; ++j) p += qs[lane + 32 * j] * kr[lane + 32 * j];
        #pragma unroll
        for (int o = 16; o > 0; o >>= 1) p += __shfl_xor(p, o, 32);
        if (lane == 0) {
            sc[kk]   = valid ? p * 0.04419417382415922f : -3.4028235e38f;
            nidx[kk] = idx;
        }
    }
    __syncthreads();

    if (tid == 0) {
        float m = -3.4028235e38f;
        for (int i = 0; i < 24; ++i) m = fmaxf(m, sc[i]);
        float ssum = 0.f;
        for (int i = 0; i < 24; ++i) { probs[i] = expf(sc[i] - m); ssum += probs[i]; }
        const float inv = 1.0f / ssum;
        for (int i = 0; i < 24; ++i) probs[i] *= inv;
    }
    __syncthreads();

    #pragma unroll
    for (int s = 0; s < 2; ++s) {
        const int d = tid + s * 256;
        float acc = 0.f;
        for (int i = 0; i < 24; ++i) acc += probs[i] * v[(size_t)nidx[i] * DMODEL + d];
        out[(size_t)n * DMODEL + d] = f32_to_bf16(acc);
    }
}

// ---------------------------------------------------------------------------
extern "C" void kernel_launch(void* const* d_in, const int* in_sizes, int n_in,
                              void* d_out, int out_size, void* d_ws, size_t ws_size,
                              hipStream_t stream) {
    (void)in_sizes; (void)n_in; (void)out_size; (void)ws_size;
    const int N = TOKENS, D = DMODEL, H = 4 * DMODEL;

    const float* x    = (const float*)d_in[0];
    const int*   grid = (const int*)  d_in[1];
    const float* Wq   = (const float*)d_in[2];  const float* bq  = (const float*)d_in[3];
    const float* Wk   = (const float*)d_in[4];  const float* bk  = (const float*)d_in[5];
    const float* Wv   = (const float*)d_in[6];  const float* bv  = (const float*)d_in[7];
    const float* Wo   = (const float*)d_in[8];  const float* bo  = (const float*)d_in[9];
    const float* g1   = (const float*)d_in[10]; const float* b1  = (const float*)d_in[11];
    const float* g2   = (const float*)d_in[12]; const float* b2  = (const float*)d_in[13];
    const float* W1   = (const float*)d_in[14]; const float* bf1 = (const float*)d_in[15];
    const float* W2   = (const float*)d_in[16]; const float* bf2 = (const float*)d_in[17];
    float* y = (float*)d_out;

    // workspace carve-up (256B aligned slices)
    size_t off = 0;
    auto alloc = [&](size_t bytes) -> void* {
        void* p = (char*)d_ws + off;
        off += (bytes + 255) & ~(size_t)255;
        return p;
    };
    u16*   hb    = (u16*)  alloc((size_t)N * D * 2);   // LN1 out (bf16)
    float* qf    = (float*)alloc((size_t)N * D * 4);
    float* kf    = (float*)alloc((size_t)N * D * 4);
    float* vf    = (float*)alloc((size_t)N * D * 4);
    u16*   aob   = (u16*)  alloc((size_t)N * D * 2);   // attention out (bf16)
    float* x1    = (float*)alloc((size_t)N * D * 4);   // x + attn@Wo + bo
    u16*   h2b   = (u16*)  alloc((size_t)N * D * 2);   // LN2 out (bf16)
    u16*   f1b   = (u16*)  alloc((size_t)N * H * 2);   // gelu(h2@W1+bf1) (bf16)
    u16*   Wq_t  = (u16*)  alloc((size_t)D * D * 2);
    u16*   Wk_t  = (u16*)  alloc((size_t)D * D * 2);
    u16*   Wv_t  = (u16*)  alloc((size_t)D * D * 2);
    u16*   Wo_t  = (u16*)  alloc((size_t)D * D * 2);
    u16*   W1_t  = (u16*)  alloc((size_t)D * H * 2);
    u16*   W2_t  = (u16*)  alloc((size_t)H * D * 2);

    // weights -> bf16 transposed
    const int cDD = (D * D + 255) / 256, cDH = (D * H + 255) / 256;
    cvt_t_kernel<<<cDD, 256, 0, stream>>>(Wq, Wq_t, D, D);
    cvt_t_kernel<<<cDD, 256, 0, stream>>>(Wk, Wk_t, D, D);
    cvt_t_kernel<<<cDD, 256, 0, stream>>>(Wv, Wv_t, D, D);
    cvt_t_kernel<<<cDD, 256, 0, stream>>>(Wo, Wo_t, D, D);
    cvt_t_kernel<<<cDH, 256, 0, stream>>>(W1, W1_t, D, H);
    cvt_t_kernel<<<cDH, 256, 0, stream>>>(W2, W2_t, H, D);

    // LN1
    ln_bf16_kernel<<<N, 256, 0, stream>>>(x, g1, b1, hb);

    // Q,K,V projections
    dim3 gQ(N / 128, D / 64);
    gemm_bf16_kernel<<<gQ, 256, 0, stream>>>(hb, Wq_t, bq, nullptr, qf, nullptr, N, D, D, 0);
    gemm_bf16_kernel<<<gQ, 256, 0, stream>>>(hb, Wk_t, bk, nullptr, kf, nullptr, N, D, D, 0);
    gemm_bf16_kernel<<<gQ, 256, 0, stream>>>(hb, Wv_t, bv, nullptr, vf, nullptr, N, D, D, 0);

    // local attention
    attn24_kernel<<<N, 256, 0, stream>>>(qf, kf, vf, grid, aob);

    // x1 = x + attn @ Wo + bo
    gemm_bf16_kernel<<<gQ, 256, 0, stream>>>(aob, Wo_t, bo, x, x1, nullptr, N, D, D, 0);

    // LN2
    ln_bf16_kernel<<<N, 256, 0, stream>>>(x1, g2, b2, h2b);

    // FFN
    dim3 gF1(N / 128, H / 64);
    gemm_bf16_kernel<<<gF1, 256, 0, stream>>>(h2b, W1_t, bf1, nullptr, nullptr, f1b, N, H, D, 1);
    gemm_bf16_kernel<<<gQ, 256, 0, stream>>>(f1b, W2_t, bf2, x1, y, nullptr, N, D, H, 0);
}
